// Bin2MirModel_55087250538634
// MI455X (gfx1250) — matrix operand order repaired and checked by reading, not compiled, not run on hardware
//
#include <hip/hip_runtime.h>
#include <hip/hip_bf16.h>
#include <math.h>

// ---------------------------------------------------------------------------
// GGNN on MI455X (gfx1250): all matmuls via v_wmma_f32_16x16x32_f16.
// Round 3: 64x64 wave tile; fully templated branch-free epilogues.
// ---------------------------------------------------------------------------

#define NNODES 4096
#define SDIM   512
#define CATDIM 1536
#define NSTEPS 5

typedef __attribute__((ext_vector_type(16))) _Float16 v16h;
typedef __attribute__((ext_vector_type(8)))  _Float16 v8h;
typedef __attribute__((ext_vector_type(8)))  float    v8f;

// ----- WMMA fragment loads (contiguous k order, same in A and B) -----
// A (16x32, MxK): lanes 0-15 -> M rows, K = 0..15; lanes 16-31 same
// rows, K = 16..31.  Two contiguous 16B chunks per lane.
__device__ __forceinline__ v16h load_a_frag(const _Float16* __restrict__ A,
                                            int lda, int m0, int k0, int lane) {
  const _Float16* p = A + (size_t)(m0 + (lane & 15)) * lda + k0 + ((lane & 16) ? 16 : 0);
  v8h lo = *(const v8h*)p;
  v8h hi = *(const v8h*)(p + 8);
  v16h r;
#pragma unroll
  for (int i = 0; i < 8; ++i) { r[i] = lo[i]; r[i + 8] = hi[i]; }
  return r;
}

// B (32x16, KxN) with B stored as rows of W (i.e. B[k][n] = W[n][k]):
// lane n<16 holds K=0..15 of column n, lane 16+n holds K=16..31.
// One contiguous 32B chunk per lane.
__device__ __forceinline__ v16h load_b_frag(const _Float16* __restrict__ W,
                                            int ldb, int n0, int k0, int lane) {
  const _Float16* p = W + (size_t)(n0 + (lane & 15)) * ldb + k0 + ((lane & 16) ? 16 : 0);
  return *(const v16h*)p;
}

// ----- Generic NT GEMM:  C[m,n] = act( sum_k A[m,k]*B[n,k] + bias )  -------
// ACT : 0 = none, 1 = sigmoid, 2 = tanh
// BIAS: 0 = none, 1 = per-row (index m), 2 = per-col (index n)
// O32 : store f32 to C ; O16 : store f16 to C (exactly one is set here)
// Block = 4 waves (2x2), wave tile 64x64 -> block tile 128x128.
// Requires M % 128 == 0, N % 128 == 0, K % 32 == 0 (true for all calls here).
template <int ACT, int BIAS, int O32, int O16>
__global__ __launch_bounds__(128) void gemm_nt_wmma(
    int K,
    const _Float16* __restrict__ A, int lda,
    const _Float16* __restrict__ B, int ldb,
    const float* __restrict__ bias,
    float* __restrict__ C32, int ldc32,
    _Float16* __restrict__ C16, int ldc16) {
  const int lane = threadIdx.x & 31;
  const int wave = threadIdx.x >> 5;
  const int mw = blockIdx.x * 128 + (wave & 1) * 64;
  const int nw = blockIdx.y * 128 + (wave >> 1) * 64;

  v8f acc[4][4] = {};
  for (int k0 = 0; k0 < K; k0 += 32) {
    v16h a[4], b[4];
#pragma unroll
    for (int i = 0; i < 4; ++i) a[i] = load_a_frag(A, lda, mw + 16 * i, k0, lane);
#pragma unroll
    for (int i = 0; i < 4; ++i) b[i] = load_b_frag(B, ldb, nw + 16 * i, k0, lane);
#pragma unroll
    for (int mi = 0; mi < 4; ++mi)
#pragma unroll
      for (int ni = 0; ni < 4; ++ni)
        acc[mi][ni] = __builtin_amdgcn_wmma_f32_16x16x32_f16(
            false, a[mi], false, b[ni], (short)0, acc[mi][ni], false, false);
  }

  // C/D layout: lanes 0-15 -> N cols, VGPR g -> M=g; lanes 16-31 -> M=8+g.
#pragma unroll
  for (int mi = 0; mi < 4; ++mi) {
#pragma unroll
    for (int ni = 0; ni < 4; ++ni) {
      const int nn = nw + ni * 16 + (lane & 15);
      const int mbase = mw + mi * 16 + ((lane & 16) ? 8 : 0);
      float bcol = 0.0f;
      if (BIAS == 2) bcol = bias[nn];
#pragma unroll
      for (int g = 0; g < 8; ++g) {
        const int mm = mbase + g;
        float v = acc[mi][ni][g];
        if (BIAS == 1) v += bias[mm];
        if (BIAS == 2) v += bcol;
        if (ACT == 1) v = 1.0f / (1.0f + __expf(-v));
        if (ACT == 2) v = tanhf(v);
        if (O32) C32[(size_t)mm * ldc32 + nn] = v;
        if (O16) C16[(size_t)mm * ldc16 + nn] = (_Float16)v;
      }
    }
  }
}

// ----------------------------- elementwise ---------------------------------
__global__ void k_cvt_f16(const float* __restrict__ s, _Float16* __restrict__ d, int n) {
  int i = blockIdx.x * 256 + threadIdx.x;
  if (i < n) d[i] = (_Float16)s[i];
}

// Convert adjacency (0/1 ints) to f16, both orientations.
__global__ void k_prep_A(const int* __restrict__ A, _Float16* __restrict__ Af,
                         _Float16* __restrict__ AfT) {
  int i = blockIdx.x * 256 + threadIdx.x;      // exact grid: NNODES*NNODES/256
  int r = i >> 12, c = i & (NNODES - 1);
  _Float16 v = (_Float16)(float)A[i];
  Af[i] = v;
  AfT[(size_t)c * NNODES + r] = v;
}

__global__ void k_init_state(const float* __restrict__ ps, float* __restrict__ s32,
                             _Float16* __restrict__ s16, int n) {
  int i = blockIdx.x * 256 + threadIdx.x;
  if (i < n) { float v = ps[i]; s32[i] = v; s16[i] = (_Float16)v; }
}

// cat[:, 1024:1536] = state
__global__ void k_cat_state(const _Float16* __restrict__ s16, _Float16* __restrict__ cat) {
  int i = blockIdx.x * 256 + threadIdx.x;      // exact grid: NNODES*SDIM/256
  int m = i >> 9, j = i & (SDIM - 1);
  cat[(size_t)m * CATDIM + 1024 + j] = s16[i];
}

// cat[:, 1024:1536] = r * state
__global__ void k_cat_rstate(const _Float16* __restrict__ r16,
                             const _Float16* __restrict__ s16,
                             _Float16* __restrict__ cat) {
  int i = blockIdx.x * 256 + threadIdx.x;
  int m = i >> 9, j = i & (SDIM - 1);
  cat[(size_t)m * CATDIM + 1024 + j] = (_Float16)((float)r16[i] * (float)s16[i]);
}

// state = (1-z)*state + z*h ; mirror to f16
__global__ void k_blend(const float* __restrict__ z, const float* __restrict__ h,
                        float* __restrict__ s32, _Float16* __restrict__ s16) {
  int i = blockIdx.x * 256 + threadIdx.x;
  float zz = z[i];
  float v = (1.0f - zz) * s32[i] + zz * h[i];
  s32[i] = v;
  s16[i] = (_Float16)v;
}

// ------------------------------- driver ------------------------------------
extern "C" void kernel_launch(void* const* d_in, const int* in_sizes, int n_in,
                              void* d_out, int out_size, void* d_ws, size_t ws_size,
                              hipStream_t stream) {
  (void)in_sizes; (void)n_in; (void)out_size; (void)ws_size;
  const float* prop_state = (const float*)d_in[0];
  const int*   Ain        = (const int*)d_in[1];
  const float* in_W  = (const float*)d_in[2];  const float* in_b  = (const float*)d_in[3];
  const float* out_W = (const float*)d_in[4];  const float* out_b = (const float*)d_in[5];
  const float* r_W   = (const float*)d_in[6];  const float* r_b   = (const float*)d_in[7];
  const float* z_W   = (const float*)d_in[8];  const float* z_b   = (const float*)d_in[9];
  const float* h_W   = (const float*)d_in[10]; const float* h_b   = (const float*)d_in[11];
  const float* o1_W  = (const float*)d_in[12]; const float* o1_b  = (const float*)d_in[13];
  const float* o2_W  = (const float*)d_in[14]; const float* o2_b  = (const float*)d_in[15];
  float* outp = (float*)d_out;

  // ---- carve workspace (256B aligned slabs) ----
  char* p = (char*)d_ws;
  auto carve = [&](size_t bytes) -> char* {
    char* r = p; p += (bytes + 255) & ~(size_t)255; return r;
  };
  const size_t NN = (size_t)NNODES * NNODES;
  const size_t NS = (size_t)NNODES * SDIM;
  _Float16* Af    = (_Float16*)carve(NN * 2);
  _Float16* AfT   = (_Float16*)carve(NN * 2);
  float*    s32   = (float*)carve(NS * 4);
  _Float16* s16   = (_Float16*)carve(NS * 2);
  _Float16* sinT  = (_Float16*)carve(NS * 2);   // [SDIM x NNODES]
  _Float16* soutT = (_Float16*)carve(NS * 2);   // [SDIM x NNODES]
  _Float16* cat   = (_Float16*)carve((size_t)NNODES * CATDIM * 2);
  _Float16* r16   = (_Float16*)carve(NS * 2);
  float*    z32   = (float*)carve(NS * 4);
  float*    h32   = (float*)carve(NS * 4);
  _Float16* t16   = (_Float16*)carve(NS * 2);
  _Float16* w_in  = (_Float16*)carve((size_t)SDIM * SDIM * 2);
  _Float16* w_out = (_Float16*)carve((size_t)SDIM * SDIM * 2);
  _Float16* w_r   = (_Float16*)carve((size_t)SDIM * CATDIM * 2);
  _Float16* w_z   = (_Float16*)carve((size_t)SDIM * CATDIM * 2);
  _Float16* w_h   = (_Float16*)carve((size_t)SDIM * CATDIM * 2);
  _Float16* w_o1  = (_Float16*)carve((size_t)SDIM * SDIM * 2);
  _Float16* w_o2  = (_Float16*)carve((size_t)SDIM * SDIM * 2);

  // ---- one-time conversions ----
  const int SQ = SDIM * SDIM, SC = SDIM * CATDIM;
  k_cvt_f16<<<(SQ + 255) / 256, 256, 0, stream>>>(in_W,  w_in,  SQ);
  k_cvt_f16<<<(SQ + 255) / 256, 256, 0, stream>>>(out_W, w_out, SQ);
  k_cvt_f16<<<(SC + 255) / 256, 256, 0, stream>>>(r_W,   w_r,   SC);
  k_cvt_f16<<<(SC + 255) / 256, 256, 0, stream>>>(z_W,   w_z,   SC);
  k_cvt_f16<<<(SC + 255) / 256, 256, 0, stream>>>(h_W,   w_h,   SC);
  k_cvt_f16<<<(SQ + 255) / 256, 256, 0, stream>>>(o1_W,  w_o1,  SQ);
  k_cvt_f16<<<(SQ + 255) / 256, 256, 0, stream>>>(o2_W,  w_o2,  SQ);
  k_prep_A<<<NN / 256, 256, 0, stream>>>(Ain, Af, AfT);
  k_init_state<<<(int)(NS / 256), 256, 0, stream>>>(prop_state, s32, s16, (int)NS);

  const dim3 blk(128);
  const dim3 g_lin(SDIM / 128, NNODES / 128);   // M=512,  N=4096 -> 4 x 32
  const dim3 g_agg(NNODES / 128, SDIM / 128);   // M=4096, N=512  -> 32 x 4
  const dim3 eblk(256);
  const int ew = (int)(NS / 256);

  for (int step = 0; step < NSTEPS; ++step) {
    // s_inT[f, node] = sum_k in_W[f,k]*state[node,k] + in_b[f]
    gemm_nt_wmma<0, 1, 0, 1><<<g_lin, blk, 0, stream>>>(SDIM,
        w_in, SDIM, s16, SDIM, in_b, nullptr, 0, sinT, NNODES);
    gemm_nt_wmma<0, 1, 0, 1><<<g_lin, blk, 0, stream>>>(SDIM,
        w_out, SDIM, s16, SDIM, out_b, nullptr, 0, soutT, NNODES);
    // a_in = A @ s_in  -> cat[:, 0:512] ; a_out = A^T @ s_out -> cat[:, 512:1024]
    gemm_nt_wmma<0, 0, 0, 1><<<g_agg, blk, 0, stream>>>(NNODES,
        Af, NNODES, sinT, NNODES, nullptr, nullptr, 0, cat, CATDIM);
    gemm_nt_wmma<0, 0, 0, 1><<<g_agg, blk, 0, stream>>>(NNODES,
        AfT, NNODES, soutT, NNODES, nullptr, nullptr, 0, cat + 512, CATDIM);
    // cat[:, 1024:1536] = state
    k_cat_state<<<ew, eblk, 0, stream>>>(s16, cat);
    // r = sigmoid(cat @ r_W^T + r_b) ; z = sigmoid(cat @ z_W^T + z_b)
    gemm_nt_wmma<1, 2, 0, 1><<<g_agg, blk, 0, stream>>>(CATDIM,
        cat, CATDIM, w_r, CATDIM, r_b, nullptr, 0, r16, SDIM);
    gemm_nt_wmma<1, 2, 1, 0><<<g_agg, blk, 0, stream>>>(CATDIM,
        cat, CATDIM, w_z, CATDIM, z_b, z32, SDIM, nullptr, 0);
    // cat[:, 1024:1536] = r * state
    k_cat_rstate<<<ew, eblk, 0, stream>>>(r16, s16, cat);
    // h_hat = tanh(cat' @ h_W^T + h_b)
    gemm_nt_wmma<2, 2, 1, 0><<<g_agg, blk, 0, stream>>>(CATDIM,
        cat, CATDIM, w_h, CATDIM, h_b, h32, SDIM, nullptr, 0);
    // state = (1-z)*state + z*h_hat
    k_blend<<<ew, eblk, 0, stream>>>(z32, h32, s32, s16);
  }

  // out = tanh(state @ o1^T + b1) @ o2^T + b2
  gemm_nt_wmma<2, 2, 0, 1><<<g_agg, blk, 0, stream>>>(SDIM,
      s16, SDIM, w_o1, SDIM, o1_b, nullptr, 0, t16, SDIM);
  gemm_nt_wmma<0, 2, 1, 0><<<g_agg, blk, 0, stream>>>(SDIM,
      t16, SDIM, w_o2, SDIM, o2_b, outp, SDIM, nullptr, 0);
}
